// CircumpunctAttention_17136919511703
// MI455X (gfx1250) — matrix operand order repaired and checked by reading, not compiled, not run on hardware
//
#include <hip/hip_runtime.h>
#include <math.h>

typedef __bf16 bf16;
typedef __bf16 v16bf __attribute__((ext_vector_type(16)));
typedef float  v8f   __attribute__((ext_vector_type(8)));
typedef unsigned int u32x4 __attribute__((ext_vector_type(4)));
typedef int i32x4 __attribute__((ext_vector_type(4)));
typedef int i32x8 __attribute__((ext_vector_type(8)));

#define T_DIM 2048
#define D_DIM 1024
#define H_DIM 16
#define DH_DIM 64
#define PI_F 3.14159265358979323846f

static __device__ inline v8f vzero8() {
  v8f z;
#pragma unroll
  for (int i = 0; i < 8; ++i) z[i] = 0.f;
  return z;
}

static __device__ inline float sigmoidf_(float v) { return 1.f / (1.f + __expf(-v)); }

// ---------------------------------------------------------------------------
// Tensor Data Mover: 2D bf16 tile load global -> LDS with per-row LDS padding.
// Builds D# group0 (count/lds_addr/global_addr/type) and group1
// (data_size, pad_enable, pad_interval, pad_amount, tile dims, dim0 stride)
// per CDNA5 ISA 8.3/8.4. Issued per-wave; caller gates to one wave per block.
// ---------------------------------------------------------------------------
static __device__ inline void tdm_load_2d_bf16(void* lds_dst, const void* gsrc,
                                               unsigned tile_w, unsigned tile_h,
                                               unsigned stride_elems,
                                               unsigned pad_interval,
                                               unsigned pad_amount) {
  const unsigned lds_addr = (unsigned)(size_t)lds_dst;  // flat addr low 32 = LDS offset
  const unsigned long long ga = (unsigned long long)(size_t)gsrc;
  u32x4 g0;
  g0[0] = 1u;                                            // count=1, user descriptor
  g0[1] = lds_addr;                                      // [63:32] lds_addr
  g0[2] = (unsigned)(ga & 0xFFFFFFFFu);                  // [95:64] global_addr lo
  g0[3] = (unsigned)((ga >> 32) & 0x01FFFFFFu) | (2u << 30);  // addr hi | type=2
  i32x8 g1;
  g1[0] = (int)((1u << 16) |                             // data_size = 2 bytes
                (1u << 20) |                             // pad_enable
                (pad_interval << 22) | (pad_amount << 25));
  const unsigned td0 = 0x40000000u;                      // huge tensor dims: no OOB
  const unsigned td1 = 0x40000000u;
  g1[1] = (int)((td0 & 0xFFFFu) << 16);                  // tensor_dim0[15:0]
  g1[2] = (int)(((td0 >> 16) & 0xFFFFu) | ((td1 & 0xFFFFu) << 16));
  g1[3] = (int)(((td1 >> 16) & 0xFFFFu) | (tile_w << 16));  // tile_dim0
  g1[4] = (int)tile_h;                                   // tile_dim1 (tile_dim2=0)
  g1[5] = (int)stride_elems;                             // tensor_dim0_stride lo32
  g1[6] = 0;
  g1[7] = 0;
  i32x4 gz;
  gz[0] = gz[1] = gz[2] = gz[3] = 0;
#if __has_include(<hip/amd_detail/amd_gfx1250_TDM.h>)
  i32x8 gz8;
#pragma unroll
  for (int i = 0; i < 8; ++i) gz8[i] = 0;
  __builtin_amdgcn_tensor_load_to_lds(g0, g1, gz, gz, gz8, 0);
#else
  __builtin_amdgcn_tensor_load_to_lds(g0, g1, gz, gz, 0);
#endif
  __builtin_amdgcn_s_wait_tensorcnt(0);
}

// ---------------------------------------------------------------------------
// fp32 -> bf16 convert
// ---------------------------------------------------------------------------
__global__ __launch_bounds__(256) void cvt_f32_bf16(const float* __restrict__ in,
                                                    bf16* __restrict__ out, size_t n) {
  size_t i = (size_t)blockIdx.x * blockDim.x + threadIdx.x;
  size_t stride = (size_t)gridDim.x * blockDim.x;
  for (; i < n; i += stride) out[i] = (bf16)in[i];
}

// ---------------------------------------------------------------------------
// aperture / expression: ap[t][h] = sigmoid(x[t]·Wa[h] + ba[h]); same for ex.
// ---------------------------------------------------------------------------
__global__ __launch_bounds__(256) void apex_kernel(const float* __restrict__ x,
                                                   const float* __restrict__ Wa,
                                                   const float* __restrict__ ba,
                                                   const float* __restrict__ We,
                                                   const float* __restrict__ be,
                                                   float* __restrict__ ap,
                                                   float* __restrict__ ex) {
  const int tid = threadIdx.x;
  const int tl = tid >> 4;
  const int h = tid & 15;
  const int t = blockIdx.x * 16 + tl;
  const float* xr = x + (size_t)t * D_DIM;
  const float* wa = Wa + (size_t)h * D_DIM;
  const float* we = We + (size_t)h * D_DIM;
  float sa = 0.f, se = 0.f;
  for (int d = 0; d < D_DIM; d += 4) {
    float4 xv = *(const float4*)(xr + d);
    float4 av = *(const float4*)(wa + d);
    float4 ev = *(const float4*)(we + d);
    sa += xv.x * av.x + xv.y * av.y + xv.z * av.z + xv.w * av.w;
    se += xv.x * ev.x + xv.y * ev.y + xv.z * ev.z + xv.w * ev.w;
  }
  ap[(size_t)t * H_DIM + h] = sigmoidf_(sa + ba[h]);
  ex[(size_t)t * H_DIM + h] = sigmoidf_(se + be[h]);
}

// ---------------------------------------------------------------------------
// C[M,N] = A[M,K] (row major, bf16) x W[N,K]^T (row major, bf16)
// Block tile 64x256, BK=32, 8 waves (2x4), wave tile 32x64 (2x4 wmma frags).
// A tile staged by TDM (wave 0); B tile transposed to [k][n] by all threads.
// ---------------------------------------------------------------------------
template <bool OUT_BF16>
__global__ __launch_bounds__(256) void gemm_xwT(const bf16* __restrict__ A,
                                                const bf16* __restrict__ W,
                                                void* __restrict__ Cout,
                                                int M, int N, int K) {
  constexpr int BM = 64, BN = 256, BK = 32;
  constexpr int SA = 40;   // padded A row stride (elems) == TDM pad 16B/row
  constexpr int SB = 264;  // padded B row stride (elems), layout [k][n]
  __shared__ __align__(16) bf16 sA[BM * SA];
  __shared__ __align__(16) bf16 sB[BK * SB];

  const int tid = threadIdx.x;
  const int lane = tid & 31;
  const int w = tid >> 5;      // 0..7
  const int wm = w >> 2;       // 0..1
  const int wn = w & 3;        // 0..3
  const int half = lane >> 4;  // 0..1
  const int l = lane & 15;

  const int m_blk = blockIdx.y * BM;
  const int n_blk = blockIdx.x * BN;

  v8f acc[2][4];
#pragma unroll
  for (int i = 0; i < 2; ++i)
#pragma unroll
    for (int j = 0; j < 4; ++j) acc[i][j] = vzero8();

  for (int k0 = 0; k0 < K; k0 += BK) {
    // --- stage A tile (64x32) via Tensor Data Mover, one wave issues
    if (w == 0)
      tdm_load_2d_bf16(sA, A + (size_t)m_blk * K + k0, BK, BM, K,
                       /*pad_interval=*/3 /* 8x8B = 64B rows */,
                       /*pad_amount=*/3 /* 4 DWORDs = 16B */);
    // --- stage B tile transposed to [k][n] (32x256): one column per thread
    {
      const int n = tid;  // 0..255
      const bf16* src = W + (size_t)(n_blk + n) * K + k0;
      bf16 tmp[32];
#pragma unroll
      for (int q = 0; q < 4; ++q) *(uint4*)(tmp + q * 8) = *(const uint4*)(src + q * 8);
#pragma unroll
      for (int j = 0; j < 32; ++j) sB[j * SB + n] = tmp[j];
    }
    __syncthreads();

    v16bf af[2], bfr[4];
#pragma unroll
    for (int fm = 0; fm < 2; ++fm) {
      const bf16* p = sA + (wm * 32 + fm * 16 + l) * SA + half * 8;
      ((uint4*)&af[fm])[0] = *(const uint4*)(p);
      ((uint4*)&af[fm])[1] = *(const uint4*)(p + 16);
    }
#pragma unroll
    for (int fn = 0; fn < 4; ++fn) {
      const bf16* p = sB + wn * 64 + fn * 16 + l;
#pragma unroll
      for (int e = 0; e < 16; ++e) bfr[fn][e] = p[(half * 8 + (e & 7) + (e >> 3) * 16) * SB];
    }
#pragma unroll
    for (int fm = 0; fm < 2; ++fm)
#pragma unroll
      for (int fn = 0; fn < 4; ++fn)
        acc[fm][fn] = __builtin_amdgcn_wmma_f32_16x16x32_bf16(
            false, af[fm], false, bfr[fn], (short)0, acc[fm][fn], false, false);
    __syncthreads();
  }

  // --- store: lane<16: M=r, N=lane ; lane>=16: M=r+8, N=lane-16
#pragma unroll
  for (int fm = 0; fm < 2; ++fm)
#pragma unroll
    for (int fn = 0; fn < 4; ++fn) {
      const int mbase = m_blk + wm * 32 + fm * 16 + half * 8;
      const int ncol = n_blk + wn * 64 + fn * 16 + l;
#pragma unroll
      for (int r = 0; r < 8; ++r) {
        if (OUT_BF16)
          ((bf16*)Cout)[(size_t)(mbase + r) * N + ncol] = (bf16)acc[fm][fn][r];
        else
          ((float*)Cout)[(size_t)(mbase + r) * N + ncol] = acc[fm][fn][r];
      }
    }
}

// ---------------------------------------------------------------------------
// Flash-style attention per head. grid = (H, T/64), block = 128 (4 waves).
// Each wave owns 16 query rows, full DH=64 accumulators.
// V chunk staged by TDM (wave 0); K chunk transposed into LDS by all threads.
// ---------------------------------------------------------------------------
__global__ __launch_bounds__(128) void attn_kernel(
    const bf16* __restrict__ inner, const bf16* __restrict__ outer,
    const bf16* __restrict__ vmat, const float* __restrict__ ap,
    const float* __restrict__ ex, const float* __restrict__ beta,
    const float* __restrict__ iv, const float* __restrict__ ov,
    const float* __restrict__ chi, bf16* __restrict__ comb) {
  const int h = blockIdx.x;
  const int qt = blockIdx.y;
  const int tid = threadIdx.x;
  const int lane = tid & 31;
  const int w = tid >> 5;      // 0..3
  const int half = lane >> 4;
  const int l = lane & 15;

  constexpr int SK = 72, SV = 72, SP = 72;
  __shared__ __align__(16) bf16 sK[64 * SK];  // [k (0..63)][s (0..63)]
  __shared__ __align__(16) bf16 sV[64 * SV];  // [s][d], SV matches TDM pad
  __shared__ __align__(16) bf16 sP[64 * SP];  // [wave-row][s]

  const float inv_scale = 0.125f;  // 1/sqrt(64)
  const int t0 = qt * 64 + w * 16;

  // Q A-fragments (K=0..31 and K=32..63) held in registers
  v16bf qa[2];
  {
    const bf16* p = inner + (size_t)(t0 + l) * D_DIM + h * DH_DIM + half * 8;
    ((uint4*)&qa[0])[0] = *(const uint4*)(p);
    ((uint4*)&qa[0])[1] = *(const uint4*)(p + 16);
    ((uint4*)&qa[1])[0] = *(const uint4*)(p + 32);
    ((uint4*)&qa[1])[1] = *(const uint4*)(p + 48);
  }

  v8f Oacc[4];
#pragma unroll
  for (int i = 0; i < 4; ++i) Oacc[i] = vzero8();
  float mrow[8], drow[8];
#pragma unroll
  for (int r = 0; r < 8; ++r) { mrow[r] = -1e30f; drow[r] = 0.f; }

  for (int s0 = 0; s0 < T_DIM; s0 += 64) {
    __syncthreads();
    // stage V chunk [s][d] via Tensor Data Mover (rows 128B, pad 16B -> SV=72)
    if (w == 0)
      tdm_load_2d_bf16(sV, vmat + (size_t)s0 * D_DIM + h * DH_DIM, DH_DIM, 64,
                       D_DIM, /*pad_interval=*/4, /*pad_amount=*/3);
    // stage K chunk transposed: sK[k][s] = outer[s0+s][h*64+k]
    {
      const int s = tid >> 1;
      const int kk = (tid & 1) * 32;
      const bf16* src = outer + (size_t)(s0 + s) * D_DIM + h * DH_DIM + kk;
      bf16 tmp[32];
#pragma unroll
      for (int q = 0; q < 4; ++q) *(uint4*)(tmp + q * 8) = *(const uint4*)(src + q * 8);
#pragma unroll
      for (int j = 0; j < 32; ++j) sK[(kk + j) * SK + s] = tmp[j];
    }
    __syncthreads();

    // scores: S = Q @ K^T  (4 n-frags x 2 k-steps)
    v8f sf[4];
#pragma unroll
    for (int n = 0; n < 4; ++n) {
      v16bf b0, b1;
      const bf16* p0 = sK + n * 16 + l;
      const bf16* p1 = sK + (size_t)32 * SK + n * 16 + l;
#pragma unroll
      for (int e = 0; e < 16; ++e) {
        const int kq = half * 8 + (e & 7) + (e >> 3) * 16;
        b0[e] = p0[kq * SK];
        b1[e] = p1[kq * SK];
      }
      v8f a = vzero8();
      a = __builtin_amdgcn_wmma_f32_16x16x32_bf16(false, qa[0], false, b0, (short)0, a, false, false);
      a = __builtin_amdgcn_wmma_f32_16x16x32_bf16(false, qa[1], false, b1, (short)0, a, false, false);
      sf[n] = a;
    }
#pragma unroll
    for (int n = 0; n < 4; ++n)
#pragma unroll
      for (int j = 0; j < 8; ++j) sf[n][j] *= inv_scale;

    // expression weights for this chunk (column = s0 + n*16 + l)
    float exv[4];
#pragma unroll
    for (int n = 0; n < 4; ++n) exv[n] = ex[(size_t)(s0 + n * 16 + l) * H_DIM + h];

    // streaming softmax update per row (row = half*8 + r within wave tile)
#pragma unroll
    for (int r = 0; r < 8; ++r) {
      float cm = -1e30f;
#pragma unroll
      for (int n = 0; n < 4; ++n) cm = fmaxf(cm, sf[n][r]);
#pragma unroll
      for (int off = 1; off < 16; off <<= 1) cm = fmaxf(cm, __shfl_xor(cm, off, 32));
      const float mnew = fmaxf(mrow[r], cm);
      const float alpha = __expf(mrow[r] - mnew);
      mrow[r] = mnew;
      drow[r] *= alpha;
#pragma unroll
      for (int nd = 0; nd < 4; ++nd) Oacc[nd][r] *= alpha;
      float rsum = 0.f;
      const int prow = w * 16 + half * 8 + r;
#pragma unroll
      for (int n = 0; n < 4; ++n) {
        const float p = __expf(sf[n][r] - mnew) * exv[n];
        rsum += p;
        sP[(size_t)prow * SP + n * 16 + l] = (bf16)p;
      }
#pragma unroll
      for (int off = 1; off < 16; off <<= 1) rsum += __shfl_xor(rsum, off, 32);
      drow[r] += rsum;
    }

    // P @ V (per-wave private sP rows; same-wave LDS ordering suffices)
    v16bf pa[2];
    {
      const bf16* p = sP + (size_t)(w * 16 + l) * SP + half * 8;
      ((uint4*)&pa[0])[0] = *(const uint4*)(p);
      ((uint4*)&pa[0])[1] = *(const uint4*)(p + 16);
      ((uint4*)&pa[1])[0] = *(const uint4*)(p + 32);
      ((uint4*)&pa[1])[1] = *(const uint4*)(p + 48);
    }
#pragma unroll
    for (int nd = 0; nd < 4; ++nd) {
      v16bf vb0, vb1;
      const bf16* p0 = sV + nd * 16 + l;
      const bf16* p1 = sV + (size_t)32 * SV + nd * 16 + l;
#pragma unroll
      for (int e = 0; e < 16; ++e) {
        const int kq = half * 8 + (e & 7) + (e >> 3) * 16;
        vb0[e] = p0[kq * SV];
        vb1[e] = p1[kq * SV];
      }
      Oacc[nd] = __builtin_amdgcn_wmma_f32_16x16x32_bf16(false, pa[0], false, vb0, (short)0, Oacc[nd], false, false);
      Oacc[nd] = __builtin_amdgcn_wmma_f32_16x16x32_bf16(false, pa[1], false, vb1, (short)0, Oacc[nd], false, false);
    }
  }

  // epilogue: aperture, renorm, complex rotation, gates, store bf16 combined
  const float ang = PI_F * sigmoidf_(beta[h]);
  const float ca = __cosf(ang), sa_ = __sinf(ang);
  const float ivs = sigmoidf_(iv[h]);
  const float og = sigmoidf_(ov[h]) * tanhf(chi[h]);

#pragma unroll
  for (int r = 0; r < 8; ++r) {
    const int t = t0 + half * 8 + r;
    const float apv = ap[(size_t)t * H_DIM + h];
    const float scale = apv / (apv * drow[r] + 1e-8f) * ivs;
    const float x0 = Oacc[0][r] * scale;
    const float x1 = Oacc[1][r] * scale;
    const float x2 = Oacc[2][r] * scale;
    const float x3 = Oacc[3][r] * scale;
    bf16* dst = comb + (size_t)t * D_DIM + h * DH_DIM + l;
    dst[0]  = (bf16)((x0 * ca - x2 * sa_) * og);
    dst[16] = (bf16)((x1 * ca - x3 * sa_) * og);
    dst[32] = (bf16)((x0 * sa_ + x2 * ca) * og);
    dst[48] = (bf16)((x1 * sa_ + x3 * ca) * og);
  }
}

// ---------------------------------------------------------------------------
extern "C" void kernel_launch(void* const* d_in, const int* in_sizes, int n_in,
                              void* d_out, int out_size, void* d_ws, size_t ws_size,
                              hipStream_t stream) {
  const float* x      = (const float*)d_in[0];
  const float* Wa_w   = (const float*)d_in[1];
  const float* Wa_b   = (const float*)d_in[2];
  const float* We_w   = (const float*)d_in[3];
  const float* We_b   = (const float*)d_in[4];
  const float* Wi_w   = (const float*)d_in[5];
  const float* Wo_w   = (const float*)d_in[6];
  const float* Wv_w   = (const float*)d_in[7];
  const float* Wout_w = (const float*)d_in[8];
  const float* beta   = (const float*)d_in[9];
  const float* iv     = (const float*)d_in[10];
  const float* ov     = (const float*)d_in[11];
  const float* chi    = (const float*)d_in[12];
  float* out = (float*)d_out;

  const size_t TD = (size_t)T_DIM * D_DIM;
  const size_t DD = (size_t)D_DIM * D_DIM;
  const size_t TH = (size_t)T_DIM * H_DIM;

  char* p = (char*)d_ws;
  bf16* xb      = (bf16*)p;  p += TD * 2;
  bf16* wi_b    = (bf16*)p;  p += DD * 2;
  bf16* wo_b    = (bf16*)p;  p += DD * 2;
  bf16* wv_b    = (bf16*)p;  p += DD * 2;
  bf16* wout_b  = (bf16*)p;  p += DD * 2;
  bf16* inner_b = (bf16*)p;  p += TD * 2;
  bf16* outer_b = (bf16*)p;  p += TD * 2;
  bf16* v_b     = (bf16*)p;  p += TD * 2;
  bf16* comb_b  = (bf16*)p;  p += TD * 2;
  float* apbuf  = (float*)p; p += TH * 4;
  float* exbuf  = (float*)p;

  cvt_f32_bf16<<<512, 256, 0, stream>>>(x, xb, TD);
  cvt_f32_bf16<<<256, 256, 0, stream>>>(Wi_w, wi_b, DD);
  cvt_f32_bf16<<<256, 256, 0, stream>>>(Wo_w, wo_b, DD);
  cvt_f32_bf16<<<256, 256, 0, stream>>>(Wv_w, wv_b, DD);
  cvt_f32_bf16<<<256, 256, 0, stream>>>(Wout_w, wout_b, DD);

  apex_kernel<<<T_DIM / 16, 256, 0, stream>>>(x, Wa_w, Wa_b, We_w, We_b, apbuf, exbuf);

  {
    dim3 grid(D_DIM / 256, T_DIM / 64);
    gemm_xwT<true><<<grid, 256, 0, stream>>>(xb, wi_b, inner_b, T_DIM, D_DIM, D_DIM);
    gemm_xwT<true><<<grid, 256, 0, stream>>>(xb, wo_b, outer_b, T_DIM, D_DIM, D_DIM);
    gemm_xwT<true><<<grid, 256, 0, stream>>>(xb, wv_b, v_b,     T_DIM, D_DIM, D_DIM);
  }

  {
    dim3 grid(H_DIM, T_DIM / 64);
    attn_kernel<<<grid, 128, 0, stream>>>(inner_b, outer_b, v_b, apbuf, exbuf,
                                          beta, iv, ov, chi, comb_b);
  }

  {
    dim3 grid(D_DIM / 256, T_DIM / 64);
    gemm_xwT<false><<<grid, 256, 0, stream>>>(comb_b, wout_b, (void*)out, T_DIM, D_DIM, D_DIM);
  }
}
